// REGNN_BIN_55405078119367
// MI455X (gfx1250) — matrix operand order repaired and checked by reading, not compiled, not run on hardware
//
#include <hip/hip_runtime.h>
#include <hip/hip_bf16.h>
#include <cstddef>

// ---------------------------------------------------------------------------
// REGNN polynomial graph filter on MI455X (gfx1250).
//
// Strategy:
//  * Convert Hm = (H+1)*0.5 to bf16 ONCE (134 MB -> fits the 192 MB L2).
//    All 40 subsequent mat-vec sweeps then stream from L2, not HBM.
//  * Mat-vec uses v_wmma_f32_16x16x32_bf16: one wave owns 16 rows, K-loop
//    step 64 (two K=32 WMMAs), B operand = vector chunk replicated across the 16 columns.
//  * Tiny elementwise / reduction kernels handle taps, leaky-relu and the
//    two mean normalizations between sweeps.
// Workspace requirement: 8192*8192*2 + ~114 KB  (~134.3 MB).
// ---------------------------------------------------------------------------

typedef __attribute__((ext_vector_type(16))) __bf16 v16bf;
typedef __attribute__((ext_vector_type(8)))  float  v8f;

#define NN 8192

__device__ __forceinline__ unsigned short f2bf(float f) {
    union { float f; unsigned u; } c; c.f = f;
    unsigned u = c.u;
    unsigned r = u + 0x7FFFu + ((u >> 16) & 1u);   // round-to-nearest-even
    return (unsigned short)(r >> 16);
}

union BfPack { uint4 q[2]; v16bf v; };

// ---------------- H conversion: Hm = (H+1)*0.5 in bf16 ---------------------
__global__ void convert_H(const float* __restrict__ H, unsigned short* __restrict__ Hb) {
    size_t i = ((size_t)blockIdx.x * blockDim.x + threadIdx.x) * 4;
    float4 h = *(const float4*)(H + i);
    ushort4 o;
    o.x = f2bf((h.x + 1.0f) * 0.5f);
    o.y = f2bf((h.y + 1.0f) * 0.5f);
    o.z = f2bf((h.z + 1.0f) * 0.5f);
    o.w = f2bf((h.w + 1.0f) * 0.5f);
    *(ushort4*)(Hb + i) = o;
}

// ---------------- x0 = (x+1)*0.5 -------------------------------------------
__global__ void init_x(const float* __restrict__ x, float* __restrict__ v,
                       unsigned short* __restrict__ vbf) {
    int i = blockIdx.x * blockDim.x + threadIdx.x;
    float t = (x[i] + 1.0f) * 0.5f;
    v[i] = t;
    vbf[i] = f2bf(t);
}

// ---------------- WMMA mat-vec: ymat = Hm(bf16) @ v(bf16) ------------------
__device__ __forceinline__ v16bf loadA(const unsigned short* __restrict__ Hb,
                                       size_t rowOff, int k0, int h) {
    BfPack u;
    const uint4* p0 = (const uint4*)(Hb + rowOff + k0 + 8 * h);        // K = 8h .. 8h+7
    const uint4* p1 = (const uint4*)(Hb + rowOff + k0 + 8 * h + 16);   // K = 8h+16 .. 8h+23
    u.q[0] = *p0;
    u.q[1] = *p1;
    return u.v;
}

__device__ __forceinline__ v16bf loadB(const unsigned short* __restrict__ vbf,
                                       int k0, int h) {
    BfPack u;
    const uint4* p = (const uint4*)(vbf + k0 + 8 * h);                 // K = 8h .. 8h+7, 8h+16 .. 8h+23
    u.q[0] = p[0];
    u.q[1] = p[2];
    return u.v;
}

__global__ void matvec_wmma(const unsigned short* __restrict__ Hb,
                            const unsigned short* __restrict__ vbf,
                            float* __restrict__ ymat) {
    const int lane    = threadIdx.x & 31;
    const int wave    = threadIdx.x >> 5;
    const int gwave   = blockIdx.x * (blockDim.x >> 5) + wave;
    const int rowBase = gwave * 16;
    const int m       = lane & 15;
    const int h       = lane >> 4;
    const size_t rowOff = (size_t)(rowBase + m) * NN;

    v8f c0 = {};
    v8f c1 = {};
    for (int k0 = 0; k0 < NN; k0 += 64) {
        v16bf a0 = loadA(Hb, rowOff, k0, h);
        v16bf b0 = loadB(vbf, k0, h);
        c0 = __builtin_amdgcn_wmma_f32_16x16x32_bf16(false, a0, false, b0,
                                                     (short)0, c0, false, false);
        v16bf a1 = loadA(Hb, rowOff, k0 + 32, h);
        v16bf b1 = loadB(vbf, k0 + 32, h);
        c1 = __builtin_amdgcn_wmma_f32_16x16x32_bf16(false, a1, false, b1,
                                                     (short)0, c1, false, false);
    }
    v8f c = c0 + c1;
    // D layout: VGPR r, lanes 0-15 -> M=r, lanes 16-31 -> M=8+r (all N columns equal).
    if (m == 0) {
#pragma unroll
        for (int r = 0; r < 8; ++r)
            ymat[rowBase + 8 * h + r] = c[r];
    }
}

// ---------------- tap j=0: yacc = alpha*v ----------------------------------
__global__ void tap0(float* __restrict__ yacc, const float* __restrict__ v,
                     const float* __restrict__ alpha, int idx) {
    int i = blockIdx.x * blockDim.x + threadIdx.x;
    yacc[i] = alpha[idx] * v[i];
}

// ---------------- tap j>0: yacc += alpha*ymat; v = ymat --------------------
__global__ void tap_acc(float* __restrict__ yacc, const float* __restrict__ ymat,
                        float* __restrict__ v, unsigned short* __restrict__ vbf,
                        const float* __restrict__ alpha, int idx) {
    int i = blockIdx.x * blockDim.x + threadIdx.x;
    float ym = ymat[i];
    yacc[i] += alpha[idx] * ym;
    v[i]   = ym;
    vbf[i] = f2bf(ym);
}

// ---------------- end of layer: leaky-relu + mean-normalize ----------------
__device__ __forceinline__ float blockReduceSum(float val, float* sbuf) {
    int t = threadIdx.x;
    sbuf[t] = val;
    __syncthreads();
    for (int s = 512; s > 0; s >>= 1) {
        if (t < s) sbuf[t] += sbuf[t + s];
        __syncthreads();
    }
    float r = sbuf[0];
    __syncthreads();
    return r;
}

__global__ void end_layer(const float* __restrict__ yacc, float* __restrict__ v,
                          unsigned short* __restrict__ vbf, float* __restrict__ out,
                          int a, int writeOut) {
    __shared__ float sbuf[1024];
    int t = threadIdx.x;
    float loc[8];
    float s = 0.0f;
#pragma unroll
    for (int e = 0; e < 8; ++e) {
        float y  = yacc[t + e * 1024];
        float lr = (y >= 0.0f) ? y : 0.01f * y;     // leaky_relu(0.01)
        loc[e] = lr;
        s += lr;
    }
    float mean1 = blockReduceSum(s, sbuf) * (1.0f / (float)NN);
    float inv   = 1.0f / mean1;
    float s2 = 0.0f;
#pragma unroll
    for (int e = 0; e < 8; ++e) { loc[e] *= inv; s2 += loc[e]; }
    float mean2 = blockReduceSum(s2, sbuf) * (1.0f / (float)NN);
#pragma unroll
    for (int e = 0; e < 8; ++e) {
        int i = t + e * 1024;
        float xo = loc[e] - mean2;
        v[i]   = xo;
        vbf[i] = f2bf(xo);
        if (writeOut)
            out[i * 2 + a] = 1.0f / (1.0f + expf(-xo));   // sigmoid, [N,2] row-major
    }
}

// ---------------------------------------------------------------------------
extern "C" void kernel_launch(void* const* d_in, const int* in_sizes, int n_in,
                              void* d_out, int out_size, void* d_ws, size_t ws_size,
                              hipStream_t stream) {
    (void)in_sizes; (void)n_in; (void)out_size; (void)ws_size;

    const float* H     = (const float*)d_in[0];   // [8192,8192] fp32
    const float* x     = (const float*)d_in[1];   // [8192,1]    fp32
    const float* alpha = (const float*)d_in[2];   // [2,5,5]     fp32
    float* out = (float*)d_out;                   // [8192,2]    fp32

    char* base = (char*)d_ws;
    unsigned short* Hb  = (unsigned short*)base;                 // 134217728 B
    size_t off = (size_t)NN * NN * 2;
    float*          v    = (float*)(base + off);          off += (size_t)NN * 4;
    unsigned short* vbf  = (unsigned short*)(base + off); off += (size_t)NN * 2;
    float*          yacc = (float*)(base + off);          off += (size_t)NN * 4;
    float*          ymat = (float*)(base + off);

    // One-time conversion (folds the (H+1)*0.5 mapping into bf16 H).
    convert_H<<<(NN / 4) * (NN / 256), 256, 0, stream>>>(H, Hb);
    init_x<<<NN / 256, 256, 0, stream>>>(x, v, vbf);

    for (int a = 0; a < 2; ++a) {
        for (int i = 0; i < 5; ++i) {
            int bi = a * 25 + i * 5;
            tap0<<<NN / 256, 256, 0, stream>>>(yacc, v, alpha, bi);
            for (int j = 1; j < 5; ++j) {
                // 512 waves, 16 rows each, full-K sweep (L2-resident bf16 H).
                matvec_wmma<<<NN / 16 / 8, 256, 0, stream>>>(Hb, vbf, ymat);
                tap_acc<<<NN / 256, 256, 0, stream>>>(yacc, ymat, v, vbf, alpha, bi + j);
            }
            end_layer<<<1, 1024, 0, stream>>>(yacc, v, vbf, out, a, (i == 4) ? 1 : 0);
        }
    }
}
